// BasicNCA_31396210933867
// MI455X (gfx1250) — matrix operand order repaired and checked by reading, not compiled, not run on hardware
//
#include <hip/hip_runtime.h>
#include <hip/hip_bf16.h>

// ---------------------------------------------------------------------------
// CDNA5 (gfx1250) NCA: per-step kernel built around v_wmma_f32_16x16x32_f16.
// wave32; A = 16 pixels x K, B = weight^T staged in LDS as f16, f32 accum.
// Round 2: removed divergent-EXEC branch around the padded K-chunk B-fragment
// (address select into a zeroed LDS pad block instead), branchless writeback.
// ---------------------------------------------------------------------------

typedef _Float16 v16h __attribute__((ext_vector_type(16)));
typedef _Float16 v8h  __attribute__((ext_vector_type(8)));
typedef float    v8f  __attribute__((ext_vector_type(8)));

#define Cc   16      // channels
#define HH   128
#define WW   128
#define TS   16      // spatial tile (16x16 pixels per workgroup)
#define HALO 18
#define KPAD 48      // y features (x|dx|dy); K padded to 64 at WMMA level
#define HID  128

__device__ __forceinline__ v8h ld8(const _Float16* p) { return *(const v8h*)p; }
__device__ __forceinline__ v16h join16(v8h lo, v8h hi) {
  return __builtin_shufflevector(lo, hi, 0,1,2,3,4,5,6,7,8,9,10,11,12,13,14,15);
}

__global__ __launch_bounds__(128)
void nca_step_kernel(const float* __restrict__ src, float* __restrict__ dst,
                     const float* __restrict__ mask_s,
                     const float* __restrict__ fc0_w, const float* __restrict__ fc0_b,
                     const float* __restrict__ fc1_w)
{
  // ~62 KB static LDS; hbuf aliases xtile after the Sobel phase.
  __shared__ __align__(16) float    xtile[HALO*HALO*Cc];  // 20736 B
  __shared__ __align__(16) _Float16 ybuf[256*KPAD];       // 24576 B
  __shared__ __align__(16) _Float16 w0s[HID*KPAD];        // 12288 B
  __shared__ __align__(16) _Float16 w1s[Cc*HID];          //  4096 B
  __shared__ __align__(16) _Float16 zpad[16];             //    32 B (all-zero)
  __shared__ float bsh[HID];
  __shared__ float msh[256];

  const int tid  = threadIdx.x;
  const int bimg = blockIdx.z;
  const int X0 = blockIdx.x * TS, Y0 = blockIdx.y * TS;
  const size_t imgbase = (size_t)bimg * HH * WW * Cc;

  // ---- phase 1: stage x tile (+halo, zero 'SAME' pad), weights(f16), bias, mask
  for (int idx = tid; idx < HALO*HALO*Cc; idx += 128) {
    int r   = idx / (HALO*Cc);
    int rem = idx - r*HALO*Cc;
    int col = rem / Cc, ch = rem - col*Cc;
    int gy = Y0 - 1 + r, gx = X0 - 1 + col;
    float v = 0.f;
    if (gy >= 0 && gy < HH && gx >= 0 && gx < WW)
      v = src[imgbase + ((size_t)gy*WW + gx)*Cc + ch];
    xtile[idx] = v;
  }
  for (int idx = tid; idx < HID*KPAD; idx += 128) w0s[idx] = (_Float16)fc0_w[idx];
  for (int idx = tid; idx < Cc*HID;  idx += 128) w1s[idx] = (_Float16)fc1_w[idx];
  if (tid < HID) bsh[tid] = fc0_b[tid];
  if (tid < 16)  zpad[tid] = (_Float16)0.f;
  for (int p = tid; p < 256; p += 128) {
    int row = p >> 4, col = p & 15;
    msh[p] = mask_s[((size_t)bimg*HH + (Y0+row))*WW + (X0+col)];
  }
  __syncthreads();

  // ---- phase 2: y = [x | sobel_x | sobel_y] in f16 (cross-correlation, /8)
  for (int p = tid; p < 256; p += 128) {
    int row = p >> 4, col = p & 15;
    #pragma unroll
    for (int ch = 0; ch < Cc; ++ch) {
      float t0 = xtile[((row  )*HALO + col  )*Cc + ch];
      float t1 = xtile[((row  )*HALO + col+1)*Cc + ch];
      float t2 = xtile[((row  )*HALO + col+2)*Cc + ch];
      float m0 = xtile[((row+1)*HALO + col  )*Cc + ch];
      float mc = xtile[((row+1)*HALO + col+1)*Cc + ch];
      float m2 = xtile[((row+1)*HALO + col+2)*Cc + ch];
      float b0 = xtile[((row+2)*HALO + col  )*Cc + ch];
      float b1 = xtile[((row+2)*HALO + col+1)*Cc + ch];
      float b2 = xtile[((row+2)*HALO + col+2)*Cc + ch];
      float dx = ((t2 + 2.f*m2 + b2) - (t0 + 2.f*m0 + b0)) * 0.125f;
      float dy = ((b0 + 2.f*b1 + b2) - (t0 + 2.f*t1 + t2)) * 0.125f;
      ybuf[p*KPAD + ch]      = (_Float16)mc;
      ybuf[p*KPAD + 16 + ch] = (_Float16)dx;
      ybuf[p*KPAD + 32 + ch] = (_Float16)dy;
    }
  }
  __syncthreads();

  // ---- phases 3+4: per-wave WMMA. One wave handles 4 tile rows (16 px each).
  const int wave = tid >> 5, lane = tid & 31;
  const int n = lane & 15, half = lane >> 4;
  const float chm = (n >= 3) ? 1.f : 0.f;                // frozen first 3 channels
  _Float16* hw = ((_Float16*)xtile) + wave * 16 * HID;   // per-wave h staging

  for (int gi = 0; gi < 4; ++gi) {
    int g = wave*4 + gi;
    // A fragments: a[i] = y[m][(i/8)*16 + half*8 + i%8 (+32)]  (ISA 16-bit A 16x32 layout)
    const _Float16* yrow = &ybuf[(g*16 + n) * KPAD];
    v8h z8 = {};
    v16h a0 = join16(ld8(yrow + half*8),      ld8(yrow + 16 + half*8));
    v16h a1 = join16(ld8(yrow + 32 + half*8), z8);  // K=48..63 are zero pad

    // fc0: h = relu(W0 y + b)  — 8 output tiles of 16, K = 32 + 16(+pad)
    #pragma unroll
    for (int t = 0; t < 8; ++t) {
      const _Float16* wrow = &w0s[(t*16 + n) * KPAD];
      // B fragment: b[i] = B[K = (i/8)*16 + half*8 + i%8][n] (same layout as A).
      v16h b0f = join16(ld8(wrow + half*8), ld8(wrow + 16 + half*8));
      // Padded chunk: upper 8 elements (K=48..63) come from a zeroed LDS block.
      v16h b1f = join16(ld8(wrow + 32 + half*8), ld8(zpad));
      float bias = bsh[t*16 + n];
      v8f acc;
      #pragma unroll
      for (int j = 0; j < 8; ++j) acc[j] = bias;
      acc = __builtin_amdgcn_wmma_f32_16x16x32_f16(false, a0, false, b0f, (short)0, acc, false, false);
      acc = __builtin_amdgcn_wmma_f32_16x16x32_f16(false, a1, false, b1f, (short)0, acc, false, false);
      // D layout: lane holds column t*16+n, rows (pixels) half*8+j -> stage h in LDS
      #pragma unroll
      for (int j = 0; j < 8; ++j)
        hw[(half*8 + j)*HID + t*16 + n] = (_Float16)fmaxf(acc[j], 0.f);
    }

    // fc1: d = h W1^T, K = 128 in 4 chunks (same-wave DS RAW is in-order)
    v8f acc2 = {};
    const _Float16* hrow  = &hw[n * HID];
    const _Float16* w1row = &w1s[n * HID];
    #pragma unroll
    for (int q = 0; q < 4; ++q) {
      v16h af = join16(ld8(hrow + q*32 + half*8),   ld8(hrow + q*32 + 16 + half*8));
      v16h bf = join16(ld8(w1row + q*32 + half*8), ld8(w1row + q*32 + 16 + half*8));
      acc2 = __builtin_amdgcn_wmma_f32_16x16x32_f16(false, af, false, bf, (short)0, acc2, false, false);
    }

    // update: lane owns channel n, columns half*8+j of tile row g (branchless)
    #pragma unroll
    for (int j = 0; j < 8; ++j) {
      int col = half*8 + j;
      size_t gidx = imgbase + ((size_t)(Y0+g)*WW + (X0+col))*Cc + n;
      float xold = src[gidx];
      dst[gidx] = fmaf(acc2[j] * chm, msh[g*16 + col], xold);
    }
  }
}

// ---- mean pool over H,W per (b,c) ------------------------------------------
__global__ __launch_bounds__(256)
void nca_pool_kernel(const float* __restrict__ xf, float* __restrict__ pooled)
{
  int b = blockIdx.x, tid = threadIdx.x;
  int c = tid & 15, sl = tid >> 4;
  const float* base = xf + (size_t)b * HH * WW * Cc;
  float s = 0.f;
  for (int p = sl; p < HH*WW; p += 16) s += base[(size_t)p*Cc + c];
  __shared__ float red[256];
  red[tid] = s; __syncthreads();
  for (int off = 128; off >= 16; off >>= 1) {
    if (tid < off) red[tid] += red[tid + off];
    __syncthreads();
  }
  if (tid < 16) pooled[b*16 + tid] = red[tid] * (1.0f / (HH*WW));
}

// ---- tiny head: relu(pooled @ fc2^T + b2) @ fc3^T + b3 ----------------------
__global__ __launch_bounds__(128)
void nca_head_kernel(const float* __restrict__ pooled,
                     const float* __restrict__ w2, const float* __restrict__ b2,
                     const float* __restrict__ w3, const float* __restrict__ b3,
                     float* __restrict__ out, int B)
{
  __shared__ float ps[64*16];
  int tid = threadIdx.x;
  for (int i = tid; i < B*16; i += 128) ps[i] = pooled[i];
  __syncthreads();
  for (int idx = tid; idx < B*13; idx += 128) {
    int b = idx / 13, k = idx - b*13;
    float acc = b3[k];
    for (int j = 0; j < 128; ++j) {
      float h = b2[j];
      #pragma unroll
      for (int c = 0; c < 16; ++c) h += ps[b*16 + c] * w2[j*16 + c];
      acc += fmaxf(h, 0.f) * w3[k*128 + j];
    }
    out[b*13 + k] = acc;
  }
}

extern "C" void kernel_launch(void* const* d_in, const int* in_sizes, int n_in,
                              void* d_out, int out_size, void* d_ws, size_t ws_size,
                              hipStream_t stream) {
  const float* x     = (const float*)d_in[0];
  const float* masks = (const float*)d_in[1];
  const float* fc0_w = (const float*)d_in[2];
  const float* fc0_b = (const float*)d_in[3];
  const float* fc1_w = (const float*)d_in[4];
  const float* fc2_w = (const float*)d_in[5];
  const float* fc2_b = (const float*)d_in[6];
  const float* fc3_w = (const float*)d_in[7];
  const float* fc3_b = (const float*)d_in[8];

  const int B     = in_sizes[0] / (HH * WW * Cc);      // 32
  const int steps = in_sizes[1] / (B * HH * WW);       // 32

  float* out = (float*)d_out;
  float* xf  = out + (size_t)B * 13;                   // second tuple element
  float* pooled = (float*)d_ws;                        // [B,16]
  float* xalt   = (float*)((char*)d_ws + 4096);        // 32 MB ping-pong buffer

  const size_t state_elems = (size_t)B * HH * WW * Cc;

  if (steps == 0) {
    hipMemcpyAsync(xf, x, state_elems * sizeof(float), hipMemcpyDeviceToDevice, stream);
  } else {
    dim3 grid(WW/TS, HH/TS, B);                        // (8, 8, B)
    const float* src = x;
    for (int s = 0; s < steps; ++s) {
      // parity chosen so the final step writes directly into d_out's xf region
      float* dst = ((s & 1) == ((steps - 1) & 1)) ? xf : xalt;
      nca_step_kernel<<<grid, 128, 0, stream>>>(src, dst,
          masks + (size_t)s * B * HH * WW, fc0_w, fc0_b, fc1_w);
      src = dst;
    }
  }
  nca_pool_kernel<<<dim3(B), 256, 0, stream>>>(xf, pooled);
  nca_head_kernel<<<dim3(1), 128, 0, stream>>>(pooled, fc2_w, fc2_b, fc3_w, fc3_b, out, B);
}
